// DifferentiableSVPF_55113020342592
// MI455X (gfx1250) — matrix operand order repaired and checked by reading, not compiled, not run on hardware
//
#include <hip/hip_runtime.h>
#include <cstdint>
#include <math.h>

#define NPART 256
#define NSTEIN 5
#define STEIN_LR 0.1f

typedef unsigned int u32x4 __attribute__((ext_vector_type(4)));
typedef int          i32x4 __attribute__((ext_vector_type(4)));
typedef int          i32x8 __attribute__((ext_vector_type(8)));
typedef _Float16     v16h  __attribute__((ext_vector_type(16)));
typedef float        v8f   __attribute__((ext_vector_type(8)));

// Pack two f32 into adjacent f16 elements of a v16h (e must be a constant
// after unrolling; cvt_pkrtz -> v_cvt_pk_f16_f32).
#define SET_PK(vec, e, x0, x1)                                   \
  {                                                              \
    auto _pk = __builtin_amdgcn_cvt_pkrtz((x0), (x1));           \
    (vec)[(e)]     = (_Float16)_pk[0];                           \
    (vec)[(e) + 1] = (_Float16)_pk[1];                           \
  }

// ---------------------------------------------------------------------------
// TDM prefetch of a contiguous row of `nelem` f32 from global into LDS.
// D# layout per CDNA5 ISA §8. TDM ignores EXEC; issue from one wave only,
// fence with s_wait_tensorcnt.
// ---------------------------------------------------------------------------
__device__ __forceinline__ void tdm_prefetch_row(const float* gsrc,
                                                 unsigned lds_byte_off,
                                                 int nelem) {
  unsigned long long ga = (unsigned long long)(uintptr_t)gsrc;
  u32x4 g0;
  g0.x = 1u;                                                   // count=1, user D#
  g0.y = lds_byte_off;                                         // lds_addr
  g0.z = (unsigned)(ga & 0xFFFFFFFFull);                       // global_addr[31:0]
  g0.w = (unsigned)((ga >> 32) & 0x01FFFFFFull) | (2u << 30);  // addr[56:32]|type=2
  i32x8 g1;
  g1[0] = (2 << 16);                         // wg_mask=0, data_size=2 (4 bytes)
  g1[1] = (nelem & 0xFFFF) << 16;            // tensor_dim0[15:0]  @ bits[63:48]
  g1[2] = ((nelem >> 16) & 0xFFFF)           // tensor_dim0[31:16] @ bits[79:64]
        | (1 << 16);                         // tensor_dim1=1      @ bits[95:80]
  g1[3] = (nelem & 0xFFFF) << 16;            // tile_dim0          @ bits[127:112]
  g1[4] = 1;                                 // tile_dim1=1, tile_dim2=0
  g1[5] = nelem;                             // tensor_dim0_stride[31:0]
  g1[6] = (nelem & 0xFFFF) << 16;            // dim0_stride[47:32]=0 | dim1_stride[15:0]
  g1[7] = 0;
  i32x4 gz = {0, 0, 0, 0};
#if defined(__clang_major__) && (__clang_major__ >= 23)
  i32x8 gz8 = {0, 0, 0, 0, 0, 0, 0, 0};
  __builtin_amdgcn_tensor_load_to_lds(g0, g1, gz, gz, gz8, 0);
#else
  __builtin_amdgcn_tensor_load_to_lds(g0, g1, gz, gz, 0);
#endif
}

__device__ __forceinline__ float wave_sum(float v) {
  #pragma unroll
  for (int o = 16; o > 0; o >>= 1) v += __shfl_down(v, o, 32);
  return v;
}

// ---------------------------------------------------------------------------
// Persistent single-workgroup SVPF scan. 256 threads = 256 particles (8 wave32
// on one WGP). Whole T-loop in-kernel (scan is serial). Stein interaction:
// per wave, 16x32 f16 tiles of K and grad_K feed two chained
// v_wmma_f32_16x16x32_f16 (C += K x G ; C += gradK x ones) so the O(n^2)
// reduction runs on the matrix pipe (co-executes with VALU element gen).
// ---------------------------------------------------------------------------
__global__ __launch_bounds__(NPART) void svpf_persistent(
    const float* __restrict__ y,
    const float* __restrict__ rho_logit_p,
    const float* __restrict__ log_sigma_z_p,
    const float* __restrict__ mu_raw_p,
    const float* __restrict__ log_nu_p,
    const float* __restrict__ eps_seq,
    const float* __restrict__ noise0,
    float* __restrict__ out, int T) {
  __shared__ __align__(16) float     hs[NPART];    // particle h (f32)
  __shared__ __align__(32) _Float16  gh[NPART];    // particle g (f16, B operand)
  __shared__ float                   phiL[NPART];  // WMMA row-sum results
  __shared__ float                   epsb[2][NPART];
  __shared__ float                   red1[8], red2[8];
  __shared__ float                   bcast[1];

  const int i    = threadIdx.x;
  const int wid  = i >> 5;
  const int lane = i & 31;
  const int rowsel = lane & 15;
  const int halfk  = (lane >= 16) ? 16 : 0;   // A K offset (contiguous k = 16h + s)
  const int bk     = (lane >= 16) ? 16 : 0;   // B-layout K offset

  // ---- uniform parameter transforms (hoisted out of the scan) ----
  const float sig      = 1.0f / (1.0f + __expf(-rho_logit_p[0]));
  const float rho      = 0.5f + 0.499f * sig;
  const float sigma_z  = fminf(fmaxf(__expf(log_sigma_z_p[0]), 0.01f), 1.0f);
  const float mu       = fminf(fmaxf(mu_raw_p[0], -10.0f), 0.0f);
  const float nu       = 2.0f + fminf(fmaxf(__expf(log_nu_p[0]), 0.1f), 100.0f);
  const float inv_sig2 = 1.0f / (sigma_z * sigma_z + 1e-8f);
  const float h_std    = sigma_z * __builtin_amdgcn_rsqf(1.0f - rho * rho + 1e-6f);
  const float Cnu = lgammaf(0.5f * (nu + 1.0f)) - lgammaf(0.5f * nu)
                  - 0.5f * logf(nu * 3.14159265358979323846f);

  // B operand of all-ones (row-sum of grad_K): 8 VGPRs of 0x3C003C00
  v16h ones;
  #pragma unroll
  for (int e = 0; e < 16; ++e) ones[e] = (_Float16)1.0f;

  float hi = mu + h_std * noise0[i];

  const unsigned eps_lds0 = (unsigned)(uintptr_t)(&epsb[0][0]);

  // Preload eps row 0 via Tensor Data Mover (wave 0 issues; per-wave op).
  if (i < 32) tdm_prefetch_row(eps_seq, eps_lds0, NPART);
  __builtin_amdgcn_s_wait_tensorcnt(0);
  __syncthreads();

  for (int t = 0; t < T; ++t) {
    const int buf = t & 1;
    // Async prefetch of eps[t+1]; completes while the Stein iterations run.
    if ((t + 1) < T && i < 32)
      tdm_prefetch_row(eps_seq + (size_t)(t + 1) * NPART,
                       eps_lds0 + (unsigned)((buf ^ 1) * NPART * 4), NPART);

    const float yt  = y[t];
    const float yt2 = yt * yt;

    // propagate particles
    const float hp = mu + rho * (hi - mu) + sigma_z * epsb[buf][i];
    hi = hp;

    // g = grad_log_p(h_pred, y_t)
    float A = yt2 / (__expf(hp) * nu + 1e-8f);
    float g = fminf(fmaxf(-(hp - mu) * inv_sig2 - 0.5f
                          + 0.5f * (nu + 1.0f) * A / (1.0f + A),
                          -10.0f), 10.0f);
    hs[i] = hp;
    gh[i] = (_Float16)g;

    // bandwidth: unbiased std(h_pred) + 0.1
    float s1 = wave_sum(hp);
    float s2 = wave_sum(hp * hp);
    if (lane == 0) { red1[wid] = s1; red2[wid] = s2; }
    __syncthreads();
    if (i == 0) {
      float S = 0.0f, Q = 0.0f;
      #pragma unroll
      for (int w = 0; w < 8; ++w) { S += red1[w]; Q += red2[w]; }
      float m   = S * (1.0f / NPART);
      float var = (Q - (float)NPART * m * m) * (1.0f / (NPART - 1));
      bcast[0] = sqrtf(fmaxf(var, 0.0f)) + 0.1f;
    }
    __syncthreads();
    const float bw      = bcast[0];
    const float inv_bw2 = 1.0f / (bw * bw);

    // ---- 5 Stein iterations; wave w owns rows [32w, 32w+32) as 2 tiles ----
    for (int s = 0; s < NSTEIN; ++s) {
      #pragma unroll
      for (int tile = 0; tile < 2; ++tile) {
        const int   i0   = (wid << 5) + (tile << 4);
        const float hrow = hs[i0 + rowsel];
        v8f c = {0.0f, 0.0f, 0.0f, 0.0f, 0.0f, 0.0f, 0.0f, 0.0f};
        for (int chunk = 0; chunk < 8; ++chunk) {
          const int jc = chunk << 5;
          // h_j for this lane's 16 A elements: one contiguous run of 16
          const float4 hA = *(const float4*)&hs[jc + halfk + 0];
          const float4 hB = *(const float4*)&hs[jc + halfk + 4];
          const float4 hC = *(const float4*)&hs[jc + halfk + 8];
          const float4 hD = *(const float4*)&hs[jc + halfk + 12];
          const float hjv[16] = {hA.x, hA.y, hA.z, hA.w, hB.x, hB.y, hB.z, hB.w,
                                 hC.x, hC.y, hC.z, hC.w, hD.x, hD.y, hD.z, hD.w};
          v16h aK, aG;
          #pragma unroll
          for (int p = 0; p < 8; ++p) {
            // element pair (2p, 2p+1): IMQ kernel + its gradient, f32
            const float d0  = hjv[2 * p]     - hrow;
            const float d1  = hjv[2 * p + 1] - hrow;
            const float dp0 = d0 * inv_bw2;
            const float dp1 = d1 * inv_bw2;
            const float b0  = fmaf(d0, dp0, 1.0f);
            const float b1  = fmaf(d1, dp1, 1.0f);
            const float r0  = __builtin_amdgcn_rsqf(b0);  // K = base^-1/2
            const float r1  = __builtin_amdgcn_rsqf(b1);
            const float gk0 = -dp0 * ((r0 * r0) * r0);    // grad_K
            const float gk1 = -dp1 * ((r1 * r1) * r1);
            SET_PK(aK, 2 * p, r0, r1);
            SET_PK(aG, 2 * p, gk0, gk1);
          }
          // B operand: g_j striped along K, identical across columns
          const v16h bG = *(const v16h*)&gh[jc + bk];
          // C += K x G  ;  C += grad_K x ones   (f32 accumulate)
          c = __builtin_amdgcn_wmma_f32_16x16x32_f16(
              false, aK, false, bG, (short)0, c, false, false);
          c = __builtin_amdgcn_wmma_f32_16x16x32_f16(
              false, aG, false, ones, (short)0, c, false, false);
        }
        // D 16x16 f32: lanes 0-15 hold rows 0-7 (v0..v7), lanes 16-31 rows
        // 8-15; columns are replicated row sums -> lanes 0,16 export 8 rows.
        if ((lane & 15) == 0) {
          const int r0w = i0 + ((lane >> 4) << 3);
          phiL[r0w + 0] = c[0]; phiL[r0w + 1] = c[1];
          phiL[r0w + 2] = c[2]; phiL[r0w + 3] = c[3];
          phiL[r0w + 4] = c[4]; phiL[r0w + 5] = c[5];
          phiL[r0w + 6] = c[6]; phiL[r0w + 7] = c[7];
        }
      }
      __syncthreads();                         // hs/gh reads done; phiL visible
      hi = fmaf(phiL[i], STEIN_LR / (float)NPART, hi);
      A = yt2 / (__expf(hi) * nu + 1e-8f);
      g = fminf(fmaxf(-(hi - mu) * inv_sig2 - 0.5f
                      + 0.5f * (nu + 1.0f) * A / (1.0f + A),
                      -10.0f), 10.0f);
      hs[i] = hi;
      gh[i] = (_Float16)g;
      __syncthreads();                         // publish updated (h, g)
    }

    // vol_mean = mean(exp(h/2)); scalar log-lik
    float sv = wave_sum(__expf(0.5f * hi));
    if (lane == 0) red1[wid] = sv;
    __syncthreads();
    if (i == 0) {
      float S = 0.0f;
      #pragma unroll
      for (int w = 0; w < 8; ++w) S += red1[w];
      float volm = S * (1.0f / NPART);
      float z    = yt / volm;
      out[t]     = volm;                                        // vols
      out[T + t] = Cnu - logf(volm)
                 - 0.5f * (nu + 1.0f) * log1pf(z * z / nu);     // lls
    }
    __builtin_amdgcn_s_wait_tensorcnt(0);  // eps[t+1] landed in LDS (wave 0)
    __syncthreads();                        // red reuse + epsb visibility
  }
}

extern "C" void kernel_launch(void* const* d_in, const int* in_sizes, int n_in,
                              void* d_out, int out_size, void* d_ws, size_t ws_size,
                              hipStream_t stream) {
  const float* y       = (const float*)d_in[0];
  const float* rho_l   = (const float*)d_in[1];
  const float* log_sz  = (const float*)d_in[2];
  const float* mu_raw  = (const float*)d_in[3];
  const float* log_nu  = (const float*)d_in[4];
  const float* eps_seq = (const float*)d_in[5];
  const float* noise0  = (const float*)d_in[6];
  float*       out     = (float*)d_out;
  const int    T       = in_sizes[0];   // 2048; n = in_sizes[6] = 256

  svpf_persistent<<<dim3(1), dim3(NPART), 0, stream>>>(
      y, rho_l, log_sz, mu_raw, log_nu, eps_seq, noise0, out, T);
}
